// BiSSMGlobalBlock_76192719831242
// MI455X (gfx1250) — matrix operand order repaired and checked by reading, not compiled, not run on hardware
//
#include <hip/hip_runtime.h>
#include <math.h>

// ---------------- problem constants ----------------
#define Bn   8
#define Nn   2048
#define Dn   768
#define Hn   768
#define Rn   4
#define Tt   16            // chunk length for parallel scan
#define Cn   (Nn / Tt)     // 128 chunks per direction
#define NBr  (Nn * Bn)     // 16384 rows for big GEMMs

typedef __bf16 bf16_t;
typedef bf16_t v16bf __attribute__((ext_vector_type(16)));
typedef bf16_t v8bf  __attribute__((ext_vector_type(8)));
typedef float  v8f   __attribute__((ext_vector_type(8)));
typedef unsigned int v4u __attribute__((ext_vector_type(4)));
typedef int v4i __attribute__((ext_vector_type(4)));
typedef int v8i __attribute__((ext_vector_type(8)));

struct TileU { union { v16bf v; v8bf h[2]; }; };

// ---- A-operand (X rows): 16x32 bf16 tile, CDNA5 A layout ----
// lane r<16: v0-3 = K0..7, v4-7 = K16..23 ; lane r+16: K8..15 / K24..31
__device__ __forceinline__ TileU load_a_tile(const bf16_t* p, int ld, int lane) {
  TileU t;
  int r  = lane & 15;
  int hh = lane >> 4;
  const bf16_t* base = p + (size_t)r * ld + hh * 8;
  t.h[0] = *(const v8bf*)(base);        // K = hh*8 .. hh*8+7
  t.h[1] = *(const v8bf*)(base + 16);   // K = 16 + hh*8 ..
  return t;
}

// ---- B-operand (W rows, computing X @ W^T): 32x16 bf16 tile, CDNA5 B layout ----
// lane n<16: K0..7, K16..23 of column n; lane n+16: K8..15 / K24..31
__device__ __forceinline__ TileU load_b_tile(const bf16_t* p, int ld, int lane) {
  TileU t;
  int n  = lane & 15;
  int ks = (lane >> 4) * 8;
  const bf16_t* base = p + (size_t)n * ld + ks;
  t.h[0] = *(const v8bf*)(base);
  t.h[1] = *(const v8bf*)(base + 16);
  return t;
}

__device__ __forceinline__ v8f wmma_bf16(const TileU& a, const TileU& b, v8f c) {
  return __builtin_amdgcn_wmma_f32_16x16x32_bf16(false, a.v, false, b.v,
                                                 (short)0, c, false, false);
}

__device__ __forceinline__ void zero_acc(v8f& a) {
#pragma unroll
  for (int i = 0; i < 8; ++i) a[i] = 0.0f;
}

// ---- TDM: 1D contiguous DMA global -> LDS, n8 elements of 8 bytes ----
// D# per CDNA5 ISA ch.8: group0 {count=1, lds_addr, global_addr[56:0], type=2},
// group1 {data_size=8B, tensor_dim0=tile_dim0=n8, tensor_dim1=tile_dim1=1, stride0=n8}
// This toolchain's builtin takes 6 args: (g0 v4u, g1 v8i, g2 v4i, g3 v4i, v8i, cpol)
__device__ __forceinline__ void tdm_load_1d_to_lds(void* lds_dst, const void* gsrc,
                                                   unsigned int n8) {
  unsigned long long ga = (unsigned long long)(uintptr_t)gsrc;
  unsigned int laddr = (unsigned int)(uintptr_t)lds_dst;   // LDS byte offset
  v4u g0;
  g0[0] = 1u;                                        // count=1 (valid user descriptor)
  g0[1] = laddr;                                     // lds_addr
  g0[2] = (unsigned int)(ga & 0xFFFFFFFFu);          // global_addr[31:0]
  g0[3] = (unsigned int)((ga >> 32) & 0x1FFFFFFu)    // global_addr[56:32]
        | (2u << 30);                                // type=2 ("image")
  v8i g1;
  g1[0] = (3 << 16);                                 // data_size = 8 bytes
  g1[1] = (int)((n8 & 0xFFFFu) << 16);               // tensor_dim0[15:0]
  g1[2] = (int)((n8 >> 16) & 0xFFFFu) | (1 << 16);   // tensor_dim0[31:16], tensor_dim1=1
  g1[3] = (int)((n8 & 0xFFFFu) << 16);               // tensor_dim1 hi=0, tile_dim0=n8
  g1[4] = 1;                                         // tile_dim1=1, tile_dim2=0
  g1[5] = (int)n8;                                   // tensor_dim0_stride[31:0]
  g1[6] = 0;                                         // stride0 hi, stride1 lo
  g1[7] = 0;
  v4i z4; z4[0] = z4[1] = z4[2] = z4[3] = 0;
  v8i z8;
#pragma unroll
  for (int i = 0; i < 8; ++i) z8[i] = 0;
  __builtin_amdgcn_tensor_load_to_lds(g0, g1, z4, z4, z8, 0);
}

// ---------------- prep: A = diag + U V^T + (S - S^T); store A (row-major) and A^T ----------------
__global__ __launch_bounds__(256) void prep_kernel(const float* __restrict__ U,
                                                   const float* __restrict__ V,
                                                   const float* __restrict__ S,
                                                   bf16_t* __restrict__ A_bf,   // PT[1] slot
                                                   bf16_t* __restrict__ At_bf) {
  int idx = blockIdx.x * 256 + threadIdx.x;     // over Hn*Hn
  int i = idx / Hn, j = idx % Hn;
  float a = (i == j) ? (-(float)(i + 1) / (float)Hn) : 0.0f;
#pragma unroll
  for (int r = 0; r < Rn; ++r) a += U[i * Rn + r] * V[j * Rn + r];
  a += S[(size_t)i * Hn + j] - S[(size_t)j * Hn + i];
  A_bf[(size_t)i * Hn + j]  = (bf16_t)a;
  At_bf[(size_t)j * Hn + i] = (bf16_t)a;
}

__global__ __launch_bounds__(256) void cvt_bf16_kernel(const float* __restrict__ in,
                                                       bf16_t* __restrict__ out) {
  int idx = blockIdx.x * 256 + threadIdx.x;
  out[idx] = (bf16_t)in[idx];
}

// ---------------- RMSNorm: x (b,n,d) -> xn bf16 stored (n*B+b, d) ----------------
__global__ __launch_bounds__(256) void rmsnorm_kernel(const float* __restrict__ x,
                                                      const float* __restrict__ scale,
                                                      bf16_t* __restrict__ xn) {
  __shared__ float red[256];
  int row = blockIdx.x;                 // b-major: b = row/N, n = row%N
  int b = row >> 11, n = row & (Nn - 1);
  const float* xr = x + (size_t)row * Dn;
  float s = 0.0f;
  for (int i = threadIdx.x; i < Dn; i += 256) { float v = xr[i]; s += v * v; }
  red[threadIdx.x] = s;
  __syncthreads();
  for (int st = 128; st > 0; st >>= 1) {
    if ((int)threadIdx.x < st) red[threadIdx.x] += red[threadIdx.x + st];
    __syncthreads();
  }
  float rinv = rsqrtf(red[0] / (float)Dn + 1e-8f);
  bf16_t* outr = xn + ((size_t)n * Bn + b) * Dn;
  for (int i = threadIdx.x; i < Dn; i += 256) outr[i] = (bf16_t)(xr[i] * rinv * scale[i]);
}

// ---------------- generic Out = X @ W^T (bf16 in, fp32 acc) ----------------
// block tile 128x128, wave tile 32x64 (2x4 WMMA tiles): 6 tile-loads per 8 WMMAs.
__global__ __launch_bounds__(256) void gemm_xwT_kernel(const bf16_t* __restrict__ X,
                                                       const bf16_t* __restrict__ W,
                                                       void* __restrict__ Out,
                                                       int Nc, int K, int out_f32) {
  int lane = threadIdx.x & 31, w = threadIdx.x >> 5;
  int mbase = blockIdx.x * 128 + (w & 3) * 32;
  int nbase = blockIdx.y * 128 + (w >> 2) * 64;
  v8f acc[2][4];
#pragma unroll
  for (int i = 0; i < 2; ++i)
#pragma unroll
    for (int j = 0; j < 4; ++j) zero_acc(acc[i][j]);
  for (int kb = 0; kb < K; kb += 32) {
    TileU a0 = load_a_tile(X + (size_t)mbase * K + kb, K, lane);
    TileU a1 = load_a_tile(X + (size_t)(mbase + 16) * K + kb, K, lane);
#pragma unroll
    for (int j = 0; j < 4; ++j) {
      TileU b = load_b_tile(W + (size_t)(nbase + j * 16) * K + kb, K, lane);
      acc[0][j] = wmma_bf16(a0, b, acc[0][j]);
      acc[1][j] = wmma_bf16(a1, b, acc[1][j]);
    }
  }
  int mo = 8 * (lane >> 4), no = lane & 15;
#pragma unroll
  for (int i = 0; i < 2; ++i)
#pragma unroll
    for (int j = 0; j < 4; ++j)
#pragma unroll
      for (int v = 0; v < 8; ++v) {
        size_t m = (size_t)mbase + i * 16 + v + mo;
        size_t n = (size_t)nbase + j * 16 + no;
        if (out_f32) ((float*)Out)[m * Nc + n] = acc[i][j][v];
        else         ((bf16_t*)Out)[m * Nc + n] = (bf16_t)acc[i][j][v];
      }
}

// ---------------- Pass A: per-chunk local scan, zero init ----------------
// State = padded 16xH bf16 in LDS (rows 8..15 stay zero).  u for the chunk is
// DMA'd into LDS by the Tensor Data Mover in two 96KB halves, overlapped with
// the first WMMA step of each half and fenced with s_wait_tensorcnt.
__global__ __launch_bounds__(256) void scan_local_kernel(const bf16_t* __restrict__ u,
                                                         const bf16_t* __restrict__ Amat,
                                                         bf16_t* __restrict__ lh) {
  __shared__ bf16_t hbuf[16 * Hn];                 // 24 KB state
  __shared__ bf16_t ubuf[(Tt / 2) * Bn * Hn];      // 96 KB staged u (half chunk)
  int tid = threadIdx.x;
  int lane = tid & 31, w = tid >> 5;
  int c = blockIdx.x;
  int dir = blockIdx.y;
  bf16_t* lhd = lh + (size_t)dir * NBr * Hn;
  // chunk's contiguous memory block starts at t0 (ascending t in memory)
  int t0 = dir ? (Nn - (c + 1) * Tt) : (c * Tt);
  const bf16_t* ublk = u + (size_t)t0 * Bn * Hn;
  const unsigned int half_elems8 = (Tt / 2) * Bn * Hn * 2 / 8;   // 8-byte units
  // phase 0 covers j=0..7: for dir=1 that's the SECOND half of the memory block
  if (w == 0)
    tdm_load_1d_to_lds(ubuf, ublk + (size_t)(dir ? 1 : 0) * (Tt / 2) * Bn * Hn, half_elems8);
  for (int i = tid; i < 16 * Hn; i += 256) hbuf[i] = (bf16_t)0.0f;
  __syncthreads();
  for (int j = 0; j < Tt; ++j) {
    int t = dir ? (Nn - 1 - (c * Tt + j)) : (c * Tt + j);
    if (j == Tt / 2 && w == 0)   // all waves are past the j=7 barrier: ubuf reusable
      tdm_load_1d_to_lds(ubuf, ublk + (size_t)(dir ? 0 : 1) * (Tt / 2) * Bn * Hn, half_elems8);
    v8f acc[6];
#pragma unroll
    for (int tt = 0; tt < 6; ++tt) zero_acc(acc[tt]);
    for (int kb = 0; kb < Hn; kb += 32) {
      TileU a = load_a_tile(hbuf + kb, Hn, lane);
#pragma unroll
      for (int tt = 0; tt < 6; ++tt) {
        int ncol = w * 96 + tt * 16;
        TileU b = load_b_tile(Amat + (size_t)ncol * Hn + kb, Hn, lane);
        acc[tt] = wmma_bf16(a, b, acc[tt]);
      }
    }
    if ((j == 0 || j == Tt / 2) && w == 0)
      __builtin_amdgcn_s_wait_tensorcnt(0);        // DMA half complete
    __syncthreads();                               // LDS reads done + u staged visible
    // u row inside the staged half: memory is ascending t, j is descending t for dir=1
    int jrow = dir ? (7 - (j & 7)) : (j & 7);
    if (lane < 16) {                               // lanes<16 hold batch rows m=0..7
#pragma unroll
      for (int tt = 0; tt < 6; ++tt) {
        int n = w * 96 + tt * 16 + lane;
#pragma unroll
        for (int v = 0; v < 8; ++v) {
          float val = acc[tt][v] + (float)ubuf[((size_t)jrow * Bn + v) * Hn + n];
          bf16_t hb = (bf16_t)val;
          hbuf[v * Hn + n] = hb;
          lhd[((size_t)t * Bn + v) * Hn + n] = hb;
        }
      }
    }
    __syncthreads();
  }
}

// ---------------- Pass B: sequential carry scan over chunks (one block per direction) ----------------
// carr[dir][c] = state entering chunk c (H_{c-1}); update carry = carry @ (A^T)^Tt + s_c
__global__ __launch_bounds__(256) void scan_carry_kernel(const bf16_t* __restrict__ lh,
                                                         const bf16_t* __restrict__ PT16,
                                                         bf16_t* __restrict__ carr) {
  __shared__ bf16_t hbuf[16 * Hn];
  int tid = threadIdx.x;
  int dir = blockIdx.x;
  const bf16_t* lhd = lh + (size_t)dir * NBr * Hn;
  bf16_t* cd = carr + (size_t)dir * Cn * Bn * Hn;
  for (int i = tid; i < 16 * Hn; i += 256) hbuf[i] = (bf16_t)0.0f;
  __syncthreads();
  int lane = tid & 31, w = tid >> 5;
  for (int c = 0; c < Cn; ++c) {
    for (int i = tid; i < Bn * Hn; i += 256) cd[(size_t)c * Bn * Hn + i] = hbuf[i];
    __syncthreads();
    if (c == Cn - 1) break;
    int tlast = dir ? (Nn - 1 - (c * Tt + Tt - 1)) : (c * Tt + Tt - 1);
    v8f acc[6];
#pragma unroll
    for (int tt = 0; tt < 6; ++tt) zero_acc(acc[tt]);
    for (int kb = 0; kb < Hn; kb += 32) {
      TileU a = load_a_tile(hbuf + kb, Hn, lane);
#pragma unroll
      for (int tt = 0; tt < 6; ++tt) {
        int ncol = w * 96 + tt * 16;
        TileU b = load_b_tile(PT16 + (size_t)ncol * Hn + kb, Hn, lane);
        acc[tt] = wmma_bf16(a, b, acc[tt]);
      }
    }
    __syncthreads();
    if (lane < 16) {
#pragma unroll
      for (int tt = 0; tt < 6; ++tt) {
        int n = w * 96 + tt * 16 + lane;
#pragma unroll
        for (int v = 0; v < 8; ++v) {
          float val = acc[tt][v] + (float)lhd[((size_t)tlast * Bn + v) * Hn + n];
          hbuf[v * Hn + n] = (bf16_t)val;
        }
      }
    }
    __syncthreads();
  }
}

// ---------------- Pass C1: h_t = carry_c @ (A^T)^{j+1} + lh_t   (in-place on lh) ----------------
__global__ __launch_bounds__(256) void broadcast_kernel(const bf16_t* __restrict__ carr,
                                                        const bf16_t* __restrict__ PTbase,
                                                        bf16_t* __restrict__ lh) {
  int t = blockIdx.x, dir = blockIdx.y;
  int p = dir ? (Nn - 1 - t) : t;
  int c = p / Tt, j = p % Tt;
  const bf16_t* cr = carr + ((size_t)dir * Cn + c) * Bn * Hn;
  const bf16_t* W  = PTbase + (size_t)(j + 1) * Hn * Hn;   // PT[j+1] = A^{j+1}
  bf16_t* lhd = lh + (size_t)dir * NBr * Hn;
  int lane = threadIdx.x & 31, w = threadIdx.x >> 5;
  v8f acc[6];
#pragma unroll
  for (int tt = 0; tt < 6; ++tt) zero_acc(acc[tt]);
  for (int kb = 0; kb < Hn; kb += 32) {
    TileU a;
    if ((lane & 15) < 8) a = load_a_tile(cr + kb, Hn, lane);   // only 8 valid rows
    else {
#pragma unroll
      for (int e = 0; e < 16; ++e) a.v[e] = (bf16_t)0.0f;
    }
#pragma unroll
    for (int tt = 0; tt < 6; ++tt) {
      int ncol = w * 96 + tt * 16;
      TileU b = load_b_tile(W + (size_t)ncol * Hn + kb, Hn, lane);
      acc[tt] = wmma_bf16(a, b, acc[tt]);
    }
  }
  if (lane < 16) {
#pragma unroll
    for (int tt = 0; tt < 6; ++tt) {
      int n = w * 96 + tt * 16 + lane;
#pragma unroll
      for (int v = 0; v < 8; ++v) {
        size_t idx = ((size_t)t * Bn + v) * Hn + n;
        float val = acc[tt][v] + (float)lhd[idx];
        lhd[idx] = (bf16_t)val;
      }
    }
  }
}

// ---------------- combine: out = x + alpha * 0.5*(yf+yb) * sigmoid(z + gate_b) ----------------
__global__ __launch_bounds__(256) void combine_kernel(const float* __restrict__ x,
                                                      const float* __restrict__ yf,
                                                      const float* __restrict__ yb,
                                                      const bf16_t* __restrict__ zg,
                                                      const float* __restrict__ gbv,
                                                      const float* __restrict__ alpha,
                                                      float* __restrict__ out) {
  size_t idx = (size_t)blockIdx.x * 256 + threadIdx.x;   // over B*N*D (b-major)
  int d = (int)(idx % Dn);
  size_t bn = idx / Dn;
  int b = (int)(bn >> 11), n = (int)(bn & (Nn - 1));
  size_t src = ((size_t)n * Bn + b) * Dn + d;
  float zv = (float)zg[src] + gbv[d];
  float g = 1.0f / (1.0f + expf(-zv));
  float y = 0.5f * (yf[src] + yb[src]) * g;
  out[idx] = x[idx] + alpha[0] * y;
}

// ---------------- host launcher ----------------
extern "C" void kernel_launch(void* const* d_in, const int* in_sizes, int n_in,
                              void* d_out, int out_size, void* d_ws, size_t ws_size,
                              hipStream_t stream) {
  const float* x     = (const float*)d_in[0];
  const float* scale = (const float*)d_in[1];
  const float* U     = (const float*)d_in[2];
  const float* V     = (const float*)d_in[3];
  const float* S     = (const float*)d_in[4];
  const float* Bm    = (const float*)d_in[5];
  const float* Cm    = (const float*)d_in[6];
  const float* gw    = (const float*)d_in[7];
  const float* gb    = (const float*)d_in[8];
  const float* alpha = (const float*)d_in[9];
  float* out = (float*)d_out;

  // workspace carve-up (~254 MB total)
  char* p = (char*)d_ws;
  auto alloc = [&](size_t bytes) -> void* {
    void* r = (void*)p;
    p += (bytes + 255) & ~(size_t)255;
    return r;
  };
  const size_t HH = (size_t)Hn * Hn;
  bf16_t* At    = (bf16_t*)alloc(HH * 2);              // A^T row-major
  bf16_t* PT    = (bf16_t*)alloc(17 * HH * 2);         // PT[k] = A^k, k=1..16 (slot 0 unused)
  bf16_t* Bm_bf = (bf16_t*)alloc(HH * 2);
  bf16_t* Cm_bf = (bf16_t*)alloc(HH * 2);
  bf16_t* gw_bf = (bf16_t*)alloc(HH * 2);
  bf16_t* xn    = (bf16_t*)alloc((size_t)NBr * Dn * 2);
  bf16_t* u     = (bf16_t*)alloc((size_t)NBr * Hn * 2);
  bf16_t* zg    = (bf16_t*)alloc((size_t)NBr * Dn * 2);
  bf16_t* lh    = (bf16_t*)alloc((size_t)2 * NBr * Hn * 2);  // fwd, bwd local/full states
  bf16_t* carr  = (bf16_t*)alloc((size_t)2 * Cn * Bn * Hn * 2);
  float*  yf    = (float*)alloc((size_t)NBr * Dn * 4);
  float*  yb    = (float*)alloc((size_t)NBr * Dn * 4);

  bf16_t* A1 = PT + HH;   // PT[1] = A

  // 1. build A / A^T
  prep_kernel<<<(Hn * Hn) / 256, 256, 0, stream>>>(U, V, S, A1, At);
  // 2. weight conversions to bf16
  cvt_bf16_kernel<<<(Hn * Dn) / 256, 256, 0, stream>>>(Bm, Bm_bf);
  cvt_bf16_kernel<<<(Dn * Hn) / 256, 256, 0, stream>>>(Cm, Cm_bf);
  cvt_bf16_kernel<<<(Dn * Dn) / 256, 256, 0, stream>>>(gw, gw_bf);
  // 3. RMSNorm -> xn (n-major rows)
  rmsnorm_kernel<<<NBr, 256, 0, stream>>>(x, scale, xn);
  // 4. u = xn @ Bm^T ; z = xn @ gate_w^T
  {
    dim3 g(NBr / 128, Hn / 128);
    gemm_xwT_kernel<<<g, 256, 0, stream>>>(xn, Bm_bf, (void*)u, Hn, Dn, 0);
    gemm_xwT_kernel<<<g, 256, 0, stream>>>(xn, gw_bf, (void*)zg, Dn, Dn, 0);
  }
  // 5. powers PT[k+1] = PT[k] @ A  (via W = A^T)
  {
    dim3 g(Hn / 128, Hn / 128);
    for (int k = 1; k < Tt; ++k)
      gemm_xwT_kernel<<<g, 256, 0, stream>>>(PT + (size_t)k * HH, At,
                                             (void*)(PT + (size_t)(k + 1) * HH), Hn, Hn, 0);
  }
  // 6. parallel local scans (128 chunks x 2 directions), u staged via TDM
  scan_local_kernel<<<dim3(Cn, 2), 256, 0, stream>>>(u, A1, lh);
  // 7. sequential carry scan (one block per direction, 127 small steps)
  scan_carry_kernel<<<2, 256, 0, stream>>>(lh, PT + (size_t)Tt * HH, carr);
  // 8. broadcast carries: lh <- carry @ A^{j+1}  + lh  (in place, parallel over all t)
  broadcast_kernel<<<dim3(Nn, 2), 256, 0, stream>>>(carr, PT, lh);
  // 9. y = h @ Cm^T (fp32 out) per direction
  {
    dim3 g(NBr / 128, Dn / 128);
    gemm_xwT_kernel<<<g, 256, 0, stream>>>(lh, Cm_bf, (void*)yf, Dn, Hn, 1);
    gemm_xwT_kernel<<<g, 256, 0, stream>>>(lh + (size_t)NBr * Hn, Cm_bf, (void*)yb, Dn, Hn, 1);
  }
  // 10. gate + skip
  combine_kernel<<<((size_t)Bn * Nn * Dn) / 256, 256, 0, stream>>>(x, yf, yb, zg, gb, alpha, out);
}
